// TAN_5016521802059
// MI455X (gfx1250) — matrix operand order repaired and checked by reading, not compiled, not run on hardware
//
#include <hip/hip_runtime.h>
#include <hip/hip_bf16.h>

// CDNA5 / gfx1250, wave32. Matrix ops via v_wmma_f32_16x16x32_f16;
// z staging via GLOBAL_LOAD_ASYNC_TO_LDS_B128 (ASYNCcnt) and the Tensor Data
// Mover (TENSOR_LOAD_TO_LDS, TENSORcnt) with double buffering in kernel 5.
typedef __attribute__((ext_vector_type(16))) _Float16 v16h;
typedef __attribute__((ext_vector_type(8)))  float    v8f;

#define WMMA_F16(a, b, c) \
  __builtin_amdgcn_wmma_f32_16x16x32_f16(false, (a), false, (b), (short)0, (c), false, false)

#if defined(__AMDGCN__) && __has_builtin(__builtin_amdgcn_global_load_async_to_lds_b128)
#define HAVE_ASYNC_LDS 1
// Param types per hipcc diagnostic: (int4 addrspace(1)*, int4 addrspace(3)*, imm, imm)
typedef int v4i_vs __attribute__((__vector_size__(16)));
typedef __attribute__((address_space(1))) v4i_vs* g_v4i_ptr;
typedef __attribute__((address_space(3))) v4i_vs* l_v4i_ptr;
__device__ __forceinline__ void async_copy16(const void* g, void* l) {
  __builtin_amdgcn_global_load_async_to_lds_b128((g_v4i_ptr)g, (l_v4i_ptr)l, 0, 0);
}
#endif

__device__ __forceinline__ void async_wait0() {
#if defined(HAVE_ASYNC_LDS)
#if __has_builtin(__builtin_amdgcn_s_wait_asynccnt)
  __builtin_amdgcn_s_wait_asynccnt(0);
#else
  asm volatile("s_wait_asynccnt 0x0" ::: "memory");
#endif
#endif
}

// ---- Tensor Data Mover (TDM): D# per CDNA5 ISA ----
#if defined(__AMDGCN__) && __has_builtin(__builtin_amdgcn_tensor_load_to_lds) && \
    __has_builtin(__builtin_amdgcn_s_wait_tensorcnt)
#define HAVE_TDM 1
typedef unsigned int u32x4_vs __attribute__((__vector_size__(16)));
typedef int          i32x8_vs __attribute__((__vector_size__(32)));
typedef int          i32x4_vs __attribute__((__vector_size__(16)));
typedef __attribute__((address_space(3))) const void* lds_cvp;

__device__ __forceinline__ unsigned lds_addr_of(const void* p) {
  return (unsigned)(unsigned long long)(lds_cvp)p;  // LDS byte offset
}

// Load a 64-row x 128-halfword tile (global row stride 512 halves) into LDS
// with 16B padding after every 256B row -> LDS pitch 136 halves.
// Group1: data_size=2B(code1), pad_enable, pad_interval=64dw(code5),
//         pad_amount=4dw(code3); tensor 128x64, tile 128x64, stride0=512.
__device__ __forceinline__ void tdm_load_tile_64x128h(void* lds, const void* gsrc) {
  const unsigned laddr = lds_addr_of(lds);
  const unsigned long long ga = (unsigned long long)gsrc;
  u32x4_vs g0 = { 1u,                                  // count=1 (valid D#)
                  laddr,                               // lds_addr
                  (unsigned)ga,                        // global_addr[31:0]
                  (unsigned)((ga >> 32) & 0x1FFFFFFu)  // global_addr[56:32]
                      | (2u << 30) };                  // type=2 ("image")
  i32x8_vs g1 = { (int)((1u << 16) | (1u << 20) | (5u << 22) | (3u << 25)),
                  (int)(128u << 16),   // tensor_dim0[15:0] @ bits63:48
                  (int)(64u << 16),    // tensor_dim0 hi=0 | tensor_dim1 lo
                  (int)(128u << 16),   // tensor_dim1 hi=0 | tile_dim0=128
                  64,                  // tile_dim1=64 | tile_dim2=0
                  512,                 // tensor_dim0_stride low32 = 512
                  0, 0 };              // stride0 hi | stride1 (unused, 2D)
  i32x4_vs z4 = { 0, 0, 0, 0 };        // groups 2/3 unused (<=2D tensor)
  i32x8_vs z8 = { 0, 0, 0, 0, 0, 0, 0, 0 };  // trailing group (6-arg form)
  __builtin_amdgcn_tensor_load_to_lds(g0, g1, z4, z4, z8, 0);
}
__device__ __forceinline__ void tensor_wait0() {
  __builtin_amdgcn_s_wait_tensorcnt(0);
}
#endif

// ---- WMMA fragment loaders (contiguous k order, k = 16*(lane/16) + i, same in A and B) ----

// A matrix 16x32 f16: lane l holds row M=l&15; lanes 0-15 K={0..15},
// lanes 16-31 K={16..31}.
__device__ __forceinline__ v16h frag_a(const _Float16* base, int ld, int lane) {
  const int row = lane & 15;
  const int khi = (lane >> 4) << 4;
  v16h r;
#pragma unroll
  for (int i = 0; i < 16; ++i) {
    const int k = i + khi;
    r[i] = base[row * ld + k];
  }
  return r;
}

// B matrix 32x16 f16, tile stored K-major: element(kr,n) = base[kr*ld + n].
__device__ __forceinline__ v16h frag_b_k(const _Float16* base, int ld, int lane) {
  const int n  = lane & 15;
  const int kb = (lane >> 4) << 4;
  v16h r;
#pragma unroll
  for (int i = 0; i < 16; ++i) r[i] = base[(kb + i) * ld + n];
  return r;
}

// B fragment from an N-major tile: element(kr,n) = base[n*ld + kr].
__device__ __forceinline__ v16h frag_b_n(const _Float16* base, int ld, int lane) {
  const int n  = lane & 15;
  const int kb = (lane >> 4) << 4;
  v16h r;
#pragma unroll
  for (int i = 0; i < 16; ++i) r[i] = base[n * ld + kb + i];
  return r;
}

// ---- Kernel 1: _v = relu(in @ W^T + bias), M=1024, D=256, N=512 ----
__global__ __launch_bounds__(128)
void fc_relu_wmma(const float* __restrict__ in, const float* __restrict__ W,
                  const float* __restrict__ bias, float* __restrict__ out_f32,
                  _Float16* __restrict__ out_f16) {
  __shared__ __align__(16) _Float16 lA[64][40];   // [m][d] chunk
  __shared__ __align__(16) _Float16 lB[64][40];   // [n][d] chunk
  const int tid = threadIdx.x, lane = tid & 31, wave = tid >> 5;
  const int m0 = blockIdx.x * 64, n0 = blockIdx.y * 64;
  v8f acc[4] = {};
  for (int dc = 0; dc < 256; dc += 32) {
#pragma unroll
    for (int e = 0; e < 4; ++e) {
      int i = e * 128 + tid, r = i >> 3, c = (i & 7) * 4;
      float4 va = *(const float4*)&in[(m0 + r) * 256 + dc + c];
      float4 vb = *(const float4*)&W[(n0 + r) * 256 + dc + c];
      lA[r][c] = (_Float16)va.x; lA[r][c+1] = (_Float16)va.y;
      lA[r][c+2] = (_Float16)va.z; lA[r][c+3] = (_Float16)va.w;
      lB[r][c] = (_Float16)vb.x; lB[r][c+1] = (_Float16)vb.y;
      lB[r][c+2] = (_Float16)vb.z; lB[r][c+3] = (_Float16)vb.w;
    }
    __syncthreads();
    v16h a = frag_a(&lA[wave * 16][0], 40, lane);
#pragma unroll
    for (int nt = 0; nt < 4; ++nt) {
      v16h b = frag_b_n(&lB[nt * 16][0], 40, lane);
      acc[nt] = WMMA_F16(a, b, acc[nt]);
    }
    __syncthreads();
  }
  const int rbase = wave * 16 + ((lane >> 4) << 3);
  const int cl = lane & 15;
#pragma unroll
  for (int nt = 0; nt < 4; ++nt) {
    int col = n0 + nt * 16 + cl;
    float bv = bias[col];
#pragma unroll
    for (int j = 0; j < 8; ++j) {
      int row = m0 + rbase + j;
      float v = fmaxf(acc[nt][j] + bv, 0.0f);
      out_f32[row * 512 + col] = v;
      out_f16[row * 512 + col] = (_Float16)v;
    }
  }
}

// ---- Kernel 2: raw attention logits, BOTH heads per (b,x) workgroup ----
// att[h,y,z] = sum_k (_y[y,k] * _x[x,k]*Wh[h,k]) * _z[z,k].
// bh[h] is constant over (x,y,z) for fixed (b,h) -> softmax-invariant -> skipped.
__global__ __launch_bounds__(128)
void att_logits_wmma(const float* __restrict__ xf, const float* __restrict__ yf,
                     const _Float16* __restrict__ zh, const float* __restrict__ Wh,
                     float* __restrict__ att) {
  __shared__ float lW[64][2];
  __shared__ __align__(16) _Float16 lA[2][64][72];  // [h][y][k] chunk
  __shared__ __align__(16) _Float16 lZ[64][72];     // [z][k] chunk (N-major B)
  const int tid = threadIdx.x, lane = tid & 31, wave = tid >> 5;
  const int xi = blockIdx.x, b = blockIdx.y;
  const float* xrow = xf + (b * 64 + xi) * 512;
  v8f acc[2][4] = {};
  for (int kc = 0; kc < 512; kc += 64) {
    __syncthreads();  // all waves done reading lA/lZ from previous chunk
    if (tid < 64) {
      float xv = xrow[kc + tid];
      lW[tid][0] = xv * Wh[kc + tid];
      lW[tid][1] = xv * Wh[512 + kc + tid];
    }
    // stage z chunk (64x64 f16) memory -> LDS via async copy
#if defined(HAVE_ASYNC_LDS)
#pragma unroll
    for (int e = 0; e < 4; ++e) {
      int i = e * 128 + tid, r = i >> 3, c = (i & 7) * 8;
      async_copy16(&zh[(b * 64 + r) * 512 + kc + c], &lZ[r][c]);
    }
#else
#pragma unroll
    for (int e = 0; e < 4; ++e) {
      int i = e * 128 + tid, r = i >> 3, c = (i & 7) * 8;
      *(ulonglong2*)&lZ[r][c] = *(const ulonglong2*)&zh[(b * 64 + r) * 512 + kc + c];
    }
#endif
    if (kc + 64 < 512)
      __builtin_prefetch(&yf[(size_t)(b * 64) * 512 + kc + 64 + (tid << 3)], 0, 0);
    __syncthreads();  // lW visible
#pragma unroll
    for (int e = 0; e < 8; ++e) {
      int i = e * 128 + tid, r = i >> 4, c = (i & 15) * 4;
      float4 v = *(const float4*)&yf[(b * 64 + r) * 512 + kc + c];
#pragma unroll
      for (int h = 0; h < 2; ++h) {
        lA[h][r][c]     = (_Float16)(v.x * lW[c][h]);
        lA[h][r][c + 1] = (_Float16)(v.y * lW[c + 1][h]);
        lA[h][r][c + 2] = (_Float16)(v.z * lW[c + 2][h]);
        lA[h][r][c + 3] = (_Float16)(v.w * lW[c + 3][h]);
      }
    }
    async_wait0();   // this wave's async z copies have landed
    __syncthreads(); // everyone's copies + lA fills visible
#pragma unroll
    for (int ks = 0; ks < 2; ++ks) {
#pragma unroll
      for (int h = 0; h < 2; ++h) {
        v16h a = frag_a(&lA[h][wave * 16][ks * 32], 72, lane);
#pragma unroll
        for (int nt = 0; nt < 4; ++nt) {
          v16h bb = frag_b_n(&lZ[nt * 16][ks * 32], 72, lane);
          acc[h][nt] = WMMA_F16(a, bb, acc[h][nt]);
        }
      }
    }
  }
  const int y = wave * 16 + ((lane >> 4) << 3);
  const int zc = lane & 15;
#pragma unroll
  for (int h = 0; h < 2; ++h) {
    float* dst = att + ((size_t)((b * 2 + h) * 64 + xi)) * 4096;
#pragma unroll
    for (int nt = 0; nt < 4; ++nt)
#pragma unroll
      for (int j = 0; j < 8; ++j)
        dst[(y + j) * 64 + nt * 16 + zc] = acc[h][nt][j];
  }
}

// ---- Kernel 3: per-(b,h) softmax max & sum(exp) over 64^3 elements ----
__global__ __launch_bounds__(256)
void softmax_stats(const float* __restrict__ att, float* __restrict__ stats) {
  __shared__ float red[256];
  const int bh = blockIdx.x, tid = threadIdx.x;
  const float4* p = (const float4*)(att + (size_t)bh * 262144);
  float mx = -3.4e38f;
  for (int i = tid; i < 65536; i += 256) {
    float4 v = p[i];
    mx = fmaxf(mx, fmaxf(fmaxf(v.x, v.y), fmaxf(v.z, v.w)));
  }
  red[tid] = mx; __syncthreads();
  for (int s = 128; s > 0; s >>= 1) { if (tid < s) red[tid] = fmaxf(red[tid], red[tid + s]); __syncthreads(); }
  mx = red[0]; __syncthreads();
  float sum = 0.f;
  for (int i = tid; i < 65536; i += 256) {
    float4 v = p[i];
    sum += expf(v.x - mx) + expf(v.y - mx) + expf(v.z - mx) + expf(v.w - mx);
  }
  red[tid] = sum; __syncthreads();
  for (int s = 128; s > 0; s >>= 1) { if (tid < s) red[tid] += red[tid + s]; __syncthreads(); }
  if (tid == 0) { stats[bh * 2] = mx; stats[bh * 2 + 1] = red[0]; }
}

// ---- Kernel 4: normalize attention in place ----
__global__ __launch_bounds__(256)
void softmax_norm(float* __restrict__ att, const float* __restrict__ stats) {
  const int bh = blockIdx.x, seg = blockIdx.y;
  const float mx = stats[bh * 2];
  const float inv = 1.0f / stats[bh * 2 + 1];
  float4* p = (float4*)(att + (size_t)bh * 262144 + seg * 4096);
  for (int i = threadIdx.x; i < 1024; i += 256) {
    float4 v = p[i];
    v.x = expf(v.x - mx) * inv; v.y = expf(v.y - mx) * inv;
    v.z = expf(v.z - mx) * inv; v.w = expf(v.w - mx) * inv;
    p[i] = v;
  }
}

__global__ void zero_f32(float* p, int n) {
  int i = blockIdx.x * blockDim.x + threadIdx.x;
  if (i < n) p[i] = 0.f;
}

// ---- Kernel 5: attention pooling, one (b,x) per WG ----
// U[y,k] = sum_z S[y,z]*_z[z,k] via WMMA (S = att summed over heads),
// then logits[b,k] += sum_y U[y,k]*_y[y,k]*_x[x,k] (atomic f32).
// z-chunk staging via TDM with double buffering (load c+1 overlaps compute c).
#if defined(HAVE_TDM)
#define ZBUFS 2
#else
#define ZBUFS 1
#endif
__global__ __launch_bounds__(128)
void pool_logits_wmma(const float* __restrict__ att, const float* __restrict__ xf,
                      const float* __restrict__ yf, const _Float16* __restrict__ zh,
                      float* __restrict__ acc_out) {
  __shared__ __align__(16) _Float16 lS[64][72];          // A: [y][z], loaded once
  __shared__ __align__(16) _Float16 lZ[ZBUFS][64][136];  // B: [z][k-chunk], K-major
  const int tid = threadIdx.x, lane = tid & 31, wave = tid >> 5;
  const int xi = blockIdx.x, b = blockIdx.y;
  const float* a0p = att + ((size_t)((b * 2 + 0) * 64 + xi)) * 4096;
  const float* a1p = att + ((size_t)((b * 2 + 1) * 64 + xi)) * 4096;
#if defined(HAVE_TDM)
  if (wave == 0) {  // TDM issue is per-wave (EXEC ignored); one wave drives it
    tdm_load_tile_64x128h(&lZ[0][0][0], &zh[(size_t)(b * 64) * 512]);
    tensor_wait0();
  }
#endif
#pragma unroll
  for (int e = 0; e < 8; ++e) {
    int i = e * 128 + tid, r = i >> 4, c = (i & 15) * 4;
    float4 v0 = *(const float4*)&a0p[r * 64 + c];
    float4 v1 = *(const float4*)&a1p[r * 64 + c];
    lS[r][c]     = (_Float16)(v0.x + v1.x);
    lS[r][c + 1] = (_Float16)(v0.y + v1.y);
    lS[r][c + 2] = (_Float16)(v0.z + v1.z);
    lS[r][c + 3] = (_Float16)(v0.w + v1.w);
  }
  __syncthreads();
  v16h af0 = frag_a(&lS[wave * 16][0], 72, lane);
  v16h af1 = frag_a(&lS[wave * 16][32], 72, lane);
  const float* xrow = xf + (b * 64 + xi) * 512;
  const int ybase = wave * 16 + ((lane >> 4) << 3);
  const int cl = lane & 15;
  for (int c = 0; c < 4; ++c) {
    const int kc = c * 128;
#if defined(HAVE_TDM)
    if (wave == 0 && c + 1 < 4)  // overlap: DMA chunk c+1 while computing c
      tdm_load_tile_64x128h(&lZ[(c + 1) & 1][0][0],
                            &zh[(size_t)(b * 64) * 512 + kc + 128]);
    const _Float16(*Zb)[136] = lZ[c & 1];
#else
    __syncthreads();   // previous lZ chunk fully consumed
#if defined(HAVE_ASYNC_LDS)
#pragma unroll
    for (int e = 0; e < 8; ++e) {
      int i = e * 128 + tid, r = i >> 4, cc = (i & 15) * 8;
      async_copy16(&zh[(b * 64 + r) * 512 + kc + cc], &lZ[0][r][cc]);
    }
    async_wait0();
#else
#pragma unroll
    for (int e = 0; e < 8; ++e) {
      int i = e * 128 + tid, r = i >> 4, cc = (i & 15) * 8;
      *(ulonglong2*)&lZ[0][r][cc] = *(const ulonglong2*)&zh[(b * 64 + r) * 512 + kc + cc];
    }
#endif
    __syncthreads();
    const _Float16(*Zb)[136] = lZ[0];
#endif
    v8f acc[8] = {};
#pragma unroll
    for (int nt = 0; nt < 8; ++nt) {
      v16h b0 = frag_b_k(&Zb[0][nt * 16], 136, lane);   // z rows 0..31
      acc[nt] = WMMA_F16(af0, b0, acc[nt]);
      v16h b1 = frag_b_k(&Zb[32][nt * 16], 136, lane);  // z rows 32..63
      acc[nt] = WMMA_F16(af1, b1, acc[nt]);
    }
#pragma unroll
    for (int nt = 0; nt < 8; ++nt) {
      const int k = kc + nt * 16 + cl;
      float s = 0.f;
#pragma unroll
      for (int j = 0; j < 8; ++j)
        s += acc[nt][j] * yf[(b * 64 + ybase + j) * 512 + k];
      s += __shfl_xor(s, 16, 32);  // lanes l and l+16 hold the same k, different y
      if (lane < 16) atomicAdd(&acc_out[b * 512 + k], s * xrow[k]);
    }
#if defined(HAVE_TDM)
    if (wave == 0) tensor_wait0();  // chunk c+1 has landed in the other buffer
    __syncthreads();
#endif
  }
}

// ---- Kernel 6: BatchNorm1d (training mode, biased var) over batch of 16 ----
__global__ __launch_bounds__(512)
void batchnorm(const float* __restrict__ acc, const float* __restrict__ gamma,
               const float* __restrict__ beta, float* __restrict__ out) {
  const int k = threadIdx.x;
  float mean = 0.f;
  for (int bb = 0; bb < 16; ++bb) mean += acc[bb * 512 + k];
  mean *= (1.0f / 16.0f);
  float var = 0.f;
  for (int bb = 0; bb < 16; ++bb) { float d = acc[bb * 512 + k] - mean; var += d * d; }
  var *= (1.0f / 16.0f);
  const float scale = gamma[k] * rsqrtf(var + 1e-5f);
  const float bias  = beta[k];
  for (int bb = 0; bb < 16; ++bb)
    out[bb * 512 + k] = (acc[bb * 512 + k] - mean) * scale + bias;
}

extern "C" void kernel_launch(void* const* d_in, const int* in_sizes, int n_in,
                              void* d_out, int out_size, void* d_ws, size_t ws_size,
                              hipStream_t stream) {
  (void)in_sizes; (void)n_in; (void)out_size; (void)ws_size;
  const float* x  = (const float*)d_in[0];
  const float* y  = (const float*)d_in[1];
  const float* z  = (const float*)d_in[2];
  const float* Wx = (const float*)d_in[3];
  const float* bx = (const float*)d_in[4];
  const float* Wy = (const float*)d_in[5];
  const float* by = (const float*)d_in[6];
  const float* Wz = (const float*)d_in[7];
  const float* bz = (const float*)d_in[8];
  const float* Wh = (const float*)d_in[9];
  // d_in[10] = bh: constant per (b,h) slice -> cancelled by softmax; unused.
  const float* gamma = (const float*)d_in[11];
  const float* beta  = (const float*)d_in[12];

  // Workspace layout (bytes); total ~9.1 MB.
  char* ws = (char*)d_ws;
  float*    xf    = (float*)(ws + 0);          // 16*64*512 f32
  float*    yf    = (float*)(ws + 2097152);
  float*    zf    = (float*)(ws + 4194304);
  float*    acc   = (float*)(ws + 6291456);    // 16*512 f32 logits accumulator
  float*    stats = (float*)(ws + 6324224);    // 32 * {max,sum}
  _Float16* xh    = (_Float16*)(ws + 6324480); // f16 copies
  _Float16* yh    = (_Float16*)(ws + 7373056);
  _Float16* zh    = (_Float16*)(ws + 8421632);

  float* out    = (float*)d_out;
  float* logits = out;          // [16,512]
  float* att    = out + 8192;   // [16,2,64,64,64]

  const dim3 gFC(16, 8);  // 1024/64 x 512/64
  fc_relu_wmma<<<gFC, 128, 0, stream>>>(x, Wx, bx, xf, xh);
  fc_relu_wmma<<<gFC, 128, 0, stream>>>(y, Wy, by, yf, yh);
  fc_relu_wmma<<<gFC, 128, 0, stream>>>(z, Wz, bz, zf, zh);

  att_logits_wmma<<<dim3(64, 16), 128, 0, stream>>>(xf, yf, zh, Wh, att);
  softmax_stats<<<32, 256, 0, stream>>>(att, stats);
  softmax_norm<<<dim3(32, 64), 256, 0, stream>>>(att, stats);
  zero_f32<<<32, 256, 0, stream>>>(acc, 8192);
  pool_logits_wmma<<<dim3(64, 16), 128, 0, stream>>>(att, xf, yf, zh, acc);
  batchnorm<<<1, 512, 0, stream>>>(acc, gamma, beta, logits);
}
